// EfficientTransformerEncoder_25683904430346
// MI455X (gfx1250) — matrix operand order repaired and checked by reading, not compiled, not run on hardware
//
#include <hip/hip_runtime.h>
#include <hip/hip_bf16.h>
#include <math.h>
#include <stdint.h>

#define EMBED    1024
#define HEADS    16
#define HEAD_DIM 64
#define FF_DIM   4096
#define N_LAYERS 4
#define BATCH    2
#define SEQ      2048
#define BSROWS   (BATCH*SEQ)
#define BAND_W   46          // ceil(sqrt(2048))

typedef __attribute__((ext_vector_type(16))) __bf16 bf16x16;
typedef __attribute__((ext_vector_type(8)))  __bf16 bf16x8;
typedef __attribute__((ext_vector_type(8)))  float  f32x8;

static __device__ __forceinline__ bf16x16 cat8(bf16x8 lo, bf16x8 hi) {
  return __builtin_shufflevector(lo, hi, 0,1,2,3,4,5,6,7,8,9,10,11,12,13,14,15);
}

// ---------------------------------------------------------------- cast f32 -> bf16
__global__ void cast_bf16_kernel(const float* __restrict__ in,
                                 __bf16* __restrict__ out, size_t n) {
  size_t i = (size_t)blockIdx.x * blockDim.x + threadIdx.x;
  if (i < n) out[i] = (__bf16)in[i];
}

// ---------------------------------------------------------------- V: [B,S,H,D] -> [B,H,D,S]
__global__ void transpose_v_kernel(const __bf16* __restrict__ in,
                                   __bf16* __restrict__ out) {
  size_t i = (size_t)blockIdx.x * 256 + threadIdx.x;
  if (i >= (size_t)BSROWS * EMBED) return;
  int e = (int)(i % EMBED);
  size_t se = i / EMBED;
  int s = (int)(se % SEQ);
  int b = (int)(se / SEQ);
  int h = e / HEAD_DIM, d = e % HEAD_DIM;
  out[(((size_t)b * HEADS + h) * HEAD_DIM + d) * SEQ + s] = in[i];
}

// ---------------------------------------------------------------- GEMM: C = act(A @ W^T + bias) [+res]
// A: [M,K] bf16 row-major, W: [N,K] bf16 row-major (torch Linear layout)
// Double-buffered LDS, tiles fetched with GLOBAL_LOAD_ASYNC_TO_LDS_B128 (ASYNCcnt).
__global__ __launch_bounds__(256)
void gemm_kernel(const __bf16* __restrict__ A, const __bf16* __restrict__ W,
                 const float* __restrict__ bias, const float* __restrict__ resid,
                 float* __restrict__ outF, __bf16* __restrict__ outB,
                 int M, int N, int K, int relu) {
  __shared__ __bf16 As[2][128][40];
  __shared__ __bf16 Bs[2][128][40];
  const int tid  = threadIdx.x;
  const int lane = tid & 31, wave = tid >> 5;
  const int hi = lane >> 4, lr = lane & 15;
  const int wm = wave >> 1, wn = wave & 1;     // 4 x 2 wave grid
  const int m0 = blockIdx.y * 128, n0 = blockIdx.x * 128;
  const int ldr = tid >> 1;                    // 0..127
  const int ldc = (tid & 1) * 16;              // 0 or 16

  const __bf16* ga0 = A + (size_t)(m0 + ldr) * K + ldc;
  const __bf16* gb0 = W + (size_t)(n0 + ldr) * K + ldc;

  // per-thread async copy of its 32B chunk of the A and B tiles for stage st / k0
  auto issue = [&](int st, int k0) {
    const __bf16* ga = ga0 + k0;
    const __bf16* gb = gb0 + k0;
    unsigned la = (unsigned)(uintptr_t)&As[st][ldr][ldc];  // LDS byte offset = addr[31:0]
    unsigned lb = (unsigned)(uintptr_t)&Bs[st][ldr][ldc];
    asm volatile(
        "global_load_async_to_lds_b128 %0, %2, off\n\t"
        "global_load_async_to_lds_b128 %0, %2, off offset:16\n\t"
        "global_load_async_to_lds_b128 %1, %3, off\n\t"
        "global_load_async_to_lds_b128 %1, %3, off offset:16"
        :: "v"(la), "v"(lb), "v"(ga), "v"(gb) : "memory");
  };

  f32x8 zero = {0.f,0.f,0.f,0.f,0.f,0.f,0.f,0.f};
  f32x8 acc[2][4];
  #pragma unroll
  for (int mt = 0; mt < 2; mt++)
    #pragma unroll
    for (int nt = 0; nt < 4; nt++) acc[mt][nt] = zero;

  issue(0, 0);
  for (int k0 = 0; k0 < K; k0 += 32) {
    const int st = (k0 >> 5) & 1;
    asm volatile("s_wait_asynccnt 0" ::: "memory");   // my 4 loads for stage st done
    __syncthreads();                                  // visible to all waves; prev reads fenced
    if (k0 + 32 < K) issue(st ^ 1, k0 + 32);          // prefetch next stage during WMMAs

    bf16x16 af[2], bfr[4];
    #pragma unroll
    for (int mt = 0; mt < 2; mt++) {
      int r = wm * 32 + mt * 16 + lr;
      bf16x8 lo = *(const bf16x8*)&As[st][r][8 * hi];
      bf16x8 hh = *(const bf16x8*)&As[st][r][16 + 8 * hi];
      af[mt] = cat8(lo, hh);
    }
    #pragma unroll
    for (int nt = 0; nt < 4; nt++) {
      int r = wn * 64 + nt * 16 + lr;
      bf16x8 lo = *(const bf16x8*)&Bs[st][r][8 * hi];
      bf16x8 hh = *(const bf16x8*)&Bs[st][r][16 + 8 * hi];
      bfr[nt] = cat8(lo, hh);
    }
    #pragma unroll
    for (int mt = 0; mt < 2; mt++)
      #pragma unroll
      for (int nt = 0; nt < 4; nt++)
        acc[mt][nt] = __builtin_amdgcn_wmma_f32_16x16x32_bf16(
            false, af[mt], false, bfr[nt], (short)0, acc[mt][nt], false, false);
  }

  #pragma unroll
  for (int mt = 0; mt < 2; mt++) {
    #pragma unroll
    for (int nt = 0; nt < 4; nt++) {
      #pragma unroll
      for (int i = 0; i < 8; i++) {
        int m = m0 + wm * 32 + mt * 16 + i + 8 * hi;
        int n = n0 + wn * 64 + nt * 16 + lr;
        float v = acc[mt][nt][i] + bias[n];
        if (relu) v = fmaxf(v, 0.f);
        size_t idx = (size_t)m * N + n;
        if (resid) v += resid[idx];
        if (outF) outF[idx] = v;
        if (outB) outB[idx] = (__bf16)v;
      }
    }
  }
}

// ---------------------------------------------------------------- band flash attention
// one wave per 16-query tile; Q,K: [B,S,E] bf16; Vt: [B,H,D,S] bf16; O: [B,S,E] bf16
__global__ __launch_bounds__(32)
void attn_kernel(const __bf16* __restrict__ Q, const __bf16* __restrict__ Kb,
                 const __bf16* __restrict__ Vt, __bf16* __restrict__ O) {
  __shared__ __bf16 P[16][40];
  const int lane = threadIdx.x;
  const int hi = lane >> 4, lr = lane & 15;
  const int tiles = SEQ / 16;
  int t = blockIdx.x;
  int qt = t % tiles; t /= tiles;
  int h = t % HEADS;
  int b = t / HEADS;

  bf16x16 qf[2];
  {
    int s = qt * 16 + lr;
    const __bf16* qrow = Q + (size_t)(b * SEQ + s) * EMBED + h * HEAD_DIM;
    #pragma unroll
    for (int kd = 0; kd < 2; kd++) {
      bf16x8 lo = *(const bf16x8*)(qrow + kd * 32 + 8 * hi);
      bf16x8 hh = *(const bf16x8*)(qrow + kd * 32 + 16 + 8 * hi);
      qf[kd] = cat8(lo, hh);
    }
  }

  f32x8 zero = {0.f,0.f,0.f,0.f,0.f,0.f,0.f,0.f};
  f32x8 oacc[4];
  float m8[8], l8[8];
  #pragma unroll
  for (int dt = 0; dt < 4; dt++) oacc[dt] = zero;
  #pragma unroll
  for (int i = 0; i < 8; i++) { m8[i] = -3.0e38f; l8[i] = 0.f; }

  const float scale = 0.125f;   // 1/sqrt(64)
  const int q0 = qt * 16;
  int jlo = q0 - BAND_W; if (jlo < 0) jlo = 0; jlo &= ~31;
  int jhi = q0 + 15 + BAND_W; if (jhi > SEQ - 1) jhi = SEQ - 1;

  for (int j0 = jlo; j0 <= jhi; j0 += 32) {
    f32x8 s0 = zero, s1 = zero;
    #pragma unroll
    for (int kd = 0; kd < 2; kd++) {
      {
        int key = j0 + lr;
        const __bf16* kr = Kb + (size_t)(b * SEQ + key) * EMBED + h * HEAD_DIM + kd * 32 + 8 * hi;
        bf16x8 lo = *(const bf16x8*)kr;
        bf16x8 hh = *(const bf16x8*)(kr + 16);
        s0 = __builtin_amdgcn_wmma_f32_16x16x32_bf16(false, qf[kd], false, cat8(lo, hh),
                                                     (short)0, s0, false, false);
      }
      {
        int key = j0 + 16 + lr;
        const __bf16* kr = Kb + (size_t)(b * SEQ + key) * EMBED + h * HEAD_DIM + kd * 32 + 8 * hi;
        bf16x8 lo = *(const bf16x8*)kr;
        bf16x8 hh = *(const bf16x8*)(kr + 16);
        s1 = __builtin_amdgcn_wmma_f32_16x16x32_bf16(false, qf[kd], false, cat8(lo, hh),
                                                     (short)0, s1, false, false);
      }
    }

    float a8[8];
    #pragma unroll
    for (int i = 0; i < 8; i++) {
      int qg = q0 + i + 8 * hi;
      int d0 = qg - (j0 + lr);       if (d0 < 0) d0 = -d0;
      int d1 = qg - (j0 + 16 + lr);  if (d1 < 0) d1 = -d1;
      float v0 = s0[i] * scale; if (d0 > BAND_W) v0 = -1e9f;
      float v1 = s1[i] * scale; if (d1 > BAND_W) v1 = -1e9f;
      float tmax = fmaxf(v0, v1);
      #pragma unroll
      for (int off = 1; off < 16; off <<= 1) tmax = fmaxf(tmax, __shfl_xor(tmax, off, 32));
      float mn = fmaxf(m8[i], tmax);
      float al = __expf(m8[i] - mn);
      float p0 = __expf(v0 - mn);
      float p1 = __expf(v1 - mn);
      float rs = p0 + p1;
      #pragma unroll
      for (int off = 1; off < 16; off <<= 1) rs += __shfl_xor(rs, off, 32);
      l8[i] = l8[i] * al + rs;
      m8[i] = mn;
      a8[i] = al;
      P[i + 8 * hi][lr]      = (__bf16)p0;
      P[i + 8 * hi][16 + lr] = (__bf16)p1;
    }
    #pragma unroll
    for (int dt = 0; dt < 4; dt++)
      #pragma unroll
      for (int i = 0; i < 8; i++) oacc[dt][i] *= a8[i];

    asm volatile("s_wait_dscnt 0" ::: "memory");  // P stores visible to cross-lane loads
    bf16x16 pf;
    {
      bf16x8 plo = *(const bf16x8*)&P[lr][8 * hi];
      bf16x8 phi = *(const bf16x8*)&P[lr][16 + 8 * hi];
      pf = cat8(plo, phi);
    }
    #pragma unroll
    for (int dt = 0; dt < 4; dt++) {
      int d = dt * 16 + lr;
      const __bf16* vr = Vt + ((size_t)(b * HEADS + h) * HEAD_DIM + d) * SEQ + j0 + 8 * hi;
      bf16x8 vlo = *(const bf16x8*)vr;
      bf16x8 vhi = *(const bf16x8*)(vr + 16);
      oacc[dt] = __builtin_amdgcn_wmma_f32_16x16x32_bf16(false, pf, false, cat8(vlo, vhi),
                                                         (short)0, oacc[dt], false, false);
    }
    asm volatile("s_wait_dscnt 0" ::: "memory");  // keep next P stores behind these loads
  }

  #pragma unroll
  for (int i = 0; i < 8; i++) {
    float inv = 1.f / l8[i];
    int s = q0 + i + 8 * hi;
    #pragma unroll
    for (int dt = 0; dt < 4; dt++) {
      float v = oacc[dt][i] * inv;
      O[(size_t)(b * SEQ + s) * EMBED + h * HEAD_DIM + dt * 16 + lr] = (__bf16)v;
    }
  }
}

// ---------------------------------------------------------------- LayerNorm (row per block)
__global__ __launch_bounds__(256)
void ln_kernel(const float* __restrict__ in, const float* __restrict__ g,
               const float* __restrict__ be, float* __restrict__ outF,
               __bf16* __restrict__ outB) {
  __shared__ float rsum[256], rsq[256];
  const int tid = threadIdx.x;
  const size_t row = blockIdx.x;
  const float* x = in + row * EMBED;
  float s1 = 0.f, s2 = 0.f;
  for (int c = tid; c < EMBED; c += 256) { float v = x[c]; s1 += v; s2 += v * v; }
  rsum[tid] = s1; rsq[tid] = s2;
  __syncthreads();
  for (int o = 128; o > 0; o >>= 1) {
    if (tid < o) { rsum[tid] += rsum[tid + o]; rsq[tid] += rsq[tid + o]; }
    __syncthreads();
  }
  float mean = rsum[0] * (1.f / EMBED);
  float var  = rsq[0] * (1.f / EMBED) - mean * mean;
  float rinv = rsqrtf(var + 1e-5f);
  for (int c = tid; c < EMBED; c += 256) {
    float v = (x[c] - mean) * rinv * g[c] + be[c];
    outF[row * EMBED + c] = v;
    outB[row * EMBED + c] = (__bf16)v;
  }
}

// ================================================================ host
extern "C" void kernel_launch(void* const* d_in, const int* in_sizes, int n_in,
                              void* d_out, int out_size, void* d_ws, size_t ws_size,
                              hipStream_t stream) {
  (void)in_sizes; (void)n_in; (void)out_size; (void)ws_size;
  const float* x   = (const float*)d_in[0];
  const float* Wq  = (const float*)d_in[1];
  const float* bq  = (const float*)d_in[2];
  const float* Wk  = (const float*)d_in[3];
  const float* bk  = (const float*)d_in[4];
  const float* Wv  = (const float*)d_in[5];
  const float* bv  = (const float*)d_in[6];
  const float* Wo  = (const float*)d_in[7];
  const float* bo  = (const float*)d_in[8];
  const float* W1  = (const float*)d_in[9];
  const float* b1  = (const float*)d_in[10];
  const float* W2  = (const float*)d_in[11];
  const float* b2  = (const float*)d_in[12];
  const float* g1  = (const float*)d_in[13];
  const float* be1 = (const float*)d_in[14];
  const float* g2  = (const float*)d_in[15];
  const float* be2 = (const float*)d_in[16];

  char* base = (char*)d_ws;
  size_t off = 0;
  auto carve = [&](size_t bytes) -> void* {
    void* r = base + off;
    off += (bytes + 255) & ~(size_t)255;
    return r;
  };

  const size_t szEE = (size_t)N_LAYERS * EMBED * EMBED;   // 4M elems
  const size_t szFE = (size_t)N_LAYERS * FF_DIM * EMBED;  // 16M elems
  const size_t szX  = (size_t)BSROWS * EMBED;             // 4M elems
  const size_t szF  = (size_t)BSROWS * FF_DIM;            // 16M elems

  __bf16* wqb  = (__bf16*)carve(szEE * 2);
  __bf16* wkb  = (__bf16*)carve(szEE * 2);
  __bf16* wvb  = (__bf16*)carve(szEE * 2);
  __bf16* wob  = (__bf16*)carve(szEE * 2);
  __bf16* w1b  = (__bf16*)carve(szFE * 2);
  __bf16* w2b  = (__bf16*)carve(szFE * 2);
  float*  xf   = (float*) carve(szX * 4);
  __bf16* xb   = (__bf16*)carve(szX * 2);
  __bf16* qb   = (__bf16*)carve(szX * 2);
  __bf16* kb   = (__bf16*)carve(szX * 2);
  __bf16* vb   = (__bf16*)carve(szX * 2);
  __bf16* vt   = (__bf16*)carve(szX * 2);
  __bf16* ob   = (__bf16*)carve(szX * 2);
  float*  x1f  = (float*) carve(szX * 4);
  __bf16* x1b  = (__bf16*)carve(szX * 2);
  __bf16* ff1b = (__bf16*)carve(szF * 2);
  float*  tmpf = (float*) carve(szX * 4);

  auto cast = [&](const float* src, __bf16* dst, size_t n) {
    cast_bf16_kernel<<<(unsigned)((n + 255) / 256), 256, 0, stream>>>(src, dst, n);
  };
  cast(Wq, wqb, szEE); cast(Wk, wkb, szEE); cast(Wv, wvb, szEE);
  cast(Wo, wob, szEE); cast(W1, w1b, szFE); cast(W2, w2b, szFE);
  cast(x, xb, szX);
  hipMemcpyAsync(xf, x, szX * 4, hipMemcpyDeviceToDevice, stream);

  dim3 gE(EMBED / 128, BSROWS / 128);     // (8, 32)
  dim3 gF(FF_DIM / 128, BSROWS / 128);    // (32, 32)
  const unsigned nAttn = BATCH * HEADS * (SEQ / 16);
  const unsigned nTr   = (unsigned)((szX + 255) / 256);

  for (int l = 0; l < N_LAYERS; l++) {
    const __bf16* wq = wqb + (size_t)l * EMBED * EMBED;
    const __bf16* wk = wkb + (size_t)l * EMBED * EMBED;
    const __bf16* wv = wvb + (size_t)l * EMBED * EMBED;
    const __bf16* wo = wob + (size_t)l * EMBED * EMBED;
    const __bf16* w1 = w1b + (size_t)l * FF_DIM * EMBED;
    const __bf16* w2 = w2b + (size_t)l * EMBED * FF_DIM;

    gemm_kernel<<<gE, 256, 0, stream>>>(xb, wq, bq + l * EMBED, nullptr,
                                        nullptr, qb, BSROWS, EMBED, EMBED, 0);
    gemm_kernel<<<gE, 256, 0, stream>>>(xb, wk, bk + l * EMBED, nullptr,
                                        nullptr, kb, BSROWS, EMBED, EMBED, 0);
    gemm_kernel<<<gE, 256, 0, stream>>>(xb, wv, bv + l * EMBED, nullptr,
                                        nullptr, vb, BSROWS, EMBED, EMBED, 0);
    transpose_v_kernel<<<nTr, 256, 0, stream>>>(vb, vt);
    attn_kernel<<<nAttn, 32, 0, stream>>>(qb, kb, vt, ob);
    gemm_kernel<<<gE, 256, 0, stream>>>(ob, wo, bo + l * EMBED, xf,
                                        tmpf, nullptr, BSROWS, EMBED, EMBED, 0);
    ln_kernel<<<BSROWS, 256, 0, stream>>>(tmpf, g1 + l * EMBED, be1 + l * EMBED, x1f, x1b);
    gemm_kernel<<<gF, 256, 0, stream>>>(x1b, w1, b1 + l * FF_DIM, nullptr,
                                        nullptr, ff1b, BSROWS, FF_DIM, EMBED, 1);
    gemm_kernel<<<gE, 256, 0, stream>>>(ff1b, w2, b2 + l * EMBED, x1f,
                                        tmpf, nullptr, BSROWS, EMBED, FF_DIM, 0);
    float* xout = (l == N_LAYERS - 1) ? (float*)d_out : xf;
    ln_kernel<<<BSROWS, 256, 0, stream>>>(tmpf, g2 + l * EMBED, be2 + l * EMBED, xout, xb);
  }
}
